// attention_head_27771258536525
// MI455X (gfx1250) — matrix operand order repaired and checked by reading, not compiled, not run on hardware
//
#include <hip/hip_runtime.h>

// ---------------------------------------------------------------------------
// MI455X (gfx1250) fused QKV+RoPE projection  +  flash attention
// - all matmuls via v_wmma_f32_16x16x32_bf16 (wave32)
// - K tiles loaded via Tensor Data Mover (tensor_load_to_lds, 6-arg form)
// - 128-bit global loads / LDS stores everywhere the layout permits
// ---------------------------------------------------------------------------

typedef __attribute__((ext_vector_type(16))) __bf16 v16bf;
typedef __attribute__((ext_vector_type(8)))  __bf16 v8bf;
typedef __attribute__((ext_vector_type(8)))  float  v8f;
typedef __attribute__((ext_vector_type(4)))  unsigned int u32x4;
typedef __attribute__((ext_vector_type(8)))  int i32x8;
typedef __attribute__((ext_vector_type(4)))  int i32x4;

#define BB   64
#define TT   512
#define CC   768
#define HH   192
#define ROT  32
#define NROW (BB * TT)   // 32768

#if defined(__has_builtin)
#if __has_builtin(__builtin_amdgcn_tensor_load_to_lds) && \
    __has_builtin(__builtin_amdgcn_s_wait_tensorcnt)
#define USE_TDM 1
#endif
#endif
#ifndef USE_TDM
#define USE_TDM 0
#endif

__device__ inline v8f wmma_bf16(v16bf a, v16bf b, v8f c) {
  return __builtin_amdgcn_wmma_f32_16x16x32_bf16(
      /*neg_a=*/false, a, /*neg_b=*/false, b,
      /*c_mod=*/(short)0, c, /*reuse_a=*/false, /*reuse_b=*/false);
}

__device__ inline v8f zero8() {
  v8f z;
#pragma unroll
  for (int i = 0; i < 8; ++i) z[i] = 0.0f;
  return z;
}

// A-fragment (16x32 bf16, MxK): lane holds row (lane&15); elements 0..15 =
// K 16*hi..16*hi+15 (contiguous, matching the B loads).
__device__ inline v16bf load_a_frag(const __bf16* row, int hi) {
  v8bf lo = *(const v8bf*)(row + 16 * hi);
  v8bf hv = *(const v8bf*)(row + 16 * hi + 8);
  v16bf a;
#pragma unroll
  for (int i = 0; i < 8; ++i) { a[i] = lo[i]; a[i + 8] = hv[i]; }
  return a;
}

// Convert 8 consecutive f32 (two float4) to a v8bf.
__device__ inline v8bf cvt8(float4 f0, float4 f1) {
  v8bf t;
  t[0] = (__bf16)f0.x; t[1] = (__bf16)f0.y; t[2] = (__bf16)f0.z; t[3] = (__bf16)f0.w;
  t[4] = (__bf16)f1.x; t[5] = (__bf16)f1.y; t[6] = (__bf16)f1.z; t[7] = (__bf16)f1.w;
  return t;
}

#if USE_TDM
// Issue a TDM 2D copy: rows x 192 bf16 tile, contiguous (stride 192), into LDS.
__device__ inline void tdm_load_k_tile(const __bf16* gsrc, unsigned int lds_off,
                                       int rows) {
  unsigned long long ga = (unsigned long long)(uintptr_t)gsrc;
  u32x4 g0;
  g0[0] = 1u;                                   // count=1, user descriptor
  g0[1] = lds_off;                              // lds_addr (bytes)
  g0[2] = (unsigned int)(ga & 0xffffffffu);     // global_addr[31:0]
  g0[3] = (unsigned int)((ga >> 32) & 0x01ffffffu) | (2u << 30);  // type=2
  i32x8 g1;
  g1[0] = (int)(1u << 16);                      // data_size=1 (2 bytes)
  g1[1] = (int)(192u << 16);                    // tensor_dim0 = 192 (bits 79:48)
  g1[2] = (int)((unsigned)rows << 16);          // tensor_dim1 = rows (bits 111:80)
  g1[3] = (int)(192u << 16);                    // tile_dim0 = 192 (bits 127:112)
  g1[4] = rows;                                 // tile_dim1 = rows (bits 143:128)
  g1[5] = 192;                                  // tensor_dim0_stride (bits 207:160)
  g1[6] = 0;
  g1[7] = 0;
  i32x4 z4;
  z4[0] = 0; z4[1] = 0; z4[2] = 0; z4[3] = 0;
  i32x8 z8;
#pragma unroll
  for (int i = 0; i < 8; ++i) z8[i] = 0;
  // 6-arg toolchain form: (g0, g1, g2, g3, extra, cpol)
  __builtin_amdgcn_tensor_load_to_lds(g0, g1, z4, z4, z8, 0);
}
#endif

// ---------------------------------------------------------------------------
// Kernel 1: q/k/v = x @ W{q,k,v}, RoPE on first 32 dims of q,k; store bf16.
// grid = (NROW/64, 3), block = 256 (8 waves). Wave -> 16x96 output stripe.
// ---------------------------------------------------------------------------
__global__ __launch_bounds__(256) void qkv_rope_kernel(
    const float* __restrict__ x,  const float* __restrict__ Wq,
    const float* __restrict__ Wk, const float* __restrict__ Wv,
    __bf16* __restrict__ ws)
{
  constexpr int XS = 40;   // x tile LDS stride (32 + pad, 80B rows)
  constexpr int WSd = 40;  // W tile (transposed) LDS stride
  __shared__ __align__(16) __bf16 xs[64 * XS];
  __shared__ __align__(16) __bf16 wt[HH * WSd];

  const int qkv = blockIdx.y;
  const float* W = (qkv == 0) ? Wq : ((qkv == 1) ? Wk : Wv);
  __bf16* dst = ws + (size_t)qkv * NROW * HH;

  const int rbase = blockIdx.x * 64;
  const int tid  = threadIdx.x;
  const int lane = tid & 31;
  const int w    = tid >> 5;
  const int hi   = lane >> 4;
  const int ln   = lane & 15;
  const int mtile = w >> 1;   // 0..3 : 16-row slice
  const int nh    = w & 1;    // 0..1 : 96-col half

  // per-thread load coordinates (hoisted out of the K loop)
  const int xr = tid >> 2;            // x tile: row 0..63
  const int xc = (tid & 3) * 8;       // x tile: col 0..31 (8-elem chunk)

  v8f acc[6];
#pragma unroll
  for (int j = 0; j < 6; ++j) acc[j] = zero8();

  for (int k0 = 0; k0 < CC; k0 += 32) {
    // x tile 64x32: 2x global_load_b128 + 1x ds_store_b128 per thread
    {
      const float4* src =
          (const float4*)&x[(size_t)(rbase + xr) * CC + (k0 + xc)];
      *(v8bf*)&xs[xr * XS + xc] = cvt8(src[0], src[1]);
    }
    // W tile 32x192 -> transposed LDS [n][k]: vector global loads,
    // scattered b16 stores on the transpose dimension only.
#pragma unroll
    for (int j = 0; j < 6; ++j) {
      int ch = tid + 256 * j;               // < 1536 chunks of 4
      int kk = ch / 48;
      int n  = (ch - kk * 48) * 4;
      float4 f = *(const float4*)&W[(size_t)(k0 + kk) * HH + n];
      wt[(n + 0) * WSd + kk] = (__bf16)f.x;
      wt[(n + 1) * WSd + kk] = (__bf16)f.y;
      wt[(n + 2) * WSd + kk] = (__bf16)f.z;
      wt[(n + 3) * WSd + kk] = (__bf16)f.w;
    }
    __syncthreads();

    v16bf a = load_a_frag(&xs[(mtile * 16 + ln) * XS], hi);
#pragma unroll
    for (int j = 0; j < 6; ++j) {
      int n = (nh * 6 + j) * 16 + ln;
      v16bf b = *(const v16bf*)&wt[n * WSd + 16 * hi];  // 16 contiguous K
      acc[j] = wmma_bf16(a, b, acc[j]);
    }
    __syncthreads();
  }

  // Epilogue: RoPE (q,k only; cols < 32) + bf16 store.
  const float ln_th_o16 = 0.57564627325f;  // ln(10000)/16
#pragma unroll
  for (int j = 0; j < 6; ++j) {
    const int ntile = nh * 6 + j;
    const int col = ntile * 16 + ln;
    const bool dorope = (qkv < 2) && (col < ROT);
    const float invf = __expf(-ln_th_o16 * (float)(col >> 1));
#pragma unroll
    for (int r = 0; r < 8; ++r) {
      const int row = rbase + mtile * 16 + r + 8 * hi;
      float v = acc[j][r];
      float partner = __shfl_xor(v, 1, 32);  // RoPE pair = adjacent lane
      float res = v;
      if (dorope) {
        float t = (float)(row & (TT - 1));
        float sn, cs;
        __sincosf(t * invf, &sn, &cs);
        res = (col & 1) ? (v * cs + partner * sn)
                        : (v * cs - partner * sn);
      }
      dst[(size_t)row * HH + col] = (__bf16)res;
    }
  }
}

// ---------------------------------------------------------------------------
// Kernel 2: causal flash attention, bf16 WMMA, f32 accumulators.
// grid = (T/128, B), block = 256 (8 waves); each wave owns 16 query rows.
// K tile via TDM (when available); V transposed into LDS manually.
// ---------------------------------------------------------------------------
__global__ __launch_bounds__(256) void flash_attn_kernel(
    const __bf16* __restrict__ qws, const __bf16* __restrict__ kws,
    const __bf16* __restrict__ vws, float* __restrict__ out)
{
  constexpr int KS = 192;  // K tile stride (contiguous: TDM copies 32x192)
  constexpr int VS = 40;   // V^T tile stride (32 keys + pad)
  constexpr int PS = 40;   // per-wave P scratch stride
  __shared__ __align__(16) __bf16 ks[32 * KS];       // [key][h]
  __shared__ __align__(16) __bf16 vt[HH * VS];       // [h][key]
  __shared__ __align__(16) __bf16 ps[8 * 16 * PS];   // per-wave [16][32]

  const int tid  = threadIdx.x;
  const int lane = tid & 31;
  const int w    = tid >> 5;
  const int hi   = lane >> 4;
  const int ln   = lane & 15;
  const int b     = blockIdx.y;
  const int qbase = blockIdx.x * 128;
  const int qr    = qbase + w * 16;   // wave's first query row (in-batch)

#if USE_TDM
  typedef __attribute__((address_space(3))) __bf16 lds_bf16;
  const unsigned int ks_off = (unsigned int)(uintptr_t)(lds_bf16*)&ks[0];
#endif

  // Preload Q fragments (16 rows x 192 dims -> 6 A-frags in registers)
  v16bf qf[6];
  {
    const __bf16* qrow = qws + ((size_t)b * TT + qr + ln) * HH;
#pragma unroll
    for (int f = 0; f < 6; ++f)
      qf[f] = load_a_frag(qrow + 32 * f, hi);
  }

  v8f o[12];
#pragma unroll
  for (int nt = 0; nt < 12; ++nt) o[nt] = zero8();
  float m[8], ls[8];
#pragma unroll
  for (int r = 0; r < 8; ++r) { m[r] = -1e30f; ls[r] = 0.0f; }

  const int nkt = blockIdx.x * 4 + 4;       // causal: enough key tiles
  const float sc = 0.07216878364f;          // 1/sqrt(192)

  for (int kt = 0; kt < nkt; ++kt) {
    const int kb = kt * 32;
    const __bf16* ktile = kws + ((size_t)b * TT + kb) * HH;
    const __bf16* vtile = vws + ((size_t)b * TT + kb) * HH;

#if USE_TDM
    if (w == 0) {
      tdm_load_k_tile(ktile, ks_off, 32);           // tensor_load_to_lds
      __builtin_amdgcn_s_wait_tensorcnt(0);
    }
#endif
    // V tile: vector global loads, transposed scatter to LDS.
    // (and K tile too when TDM is unavailable)
#pragma unroll
    for (int j = 0; j < 3; ++j) {
      int ch  = tid + 256 * j;                     // < 768 chunks of 8
      int key = ch / 24;
      int h   = (ch - key * 24) * 8;
      v8bf vv = *(const v8bf*)&vtile[(size_t)key * HH + h];
#pragma unroll
      for (int i = 0; i < 8; ++i) vt[(h + i) * VS + key] = vv[i];
#if !USE_TDM
      *(v8bf*)&ks[key * KS + h] = *(const v8bf*)&ktile[(size_t)key * HH + h];
#endif
    }
    if (kt + 1 < nkt)  // global_prefetch_b8 for next K tile
      __builtin_prefetch(&ktile[(size_t)32 * HH], 0, 0);
    __syncthreads();

    // S = Q . K^T  (two 16x16 tiles: keys [kb, kb+16) and [kb+16, kb+32))
    v8f s0 = zero8(), s1 = zero8();
#pragma unroll
    for (int f = 0; f < 6; ++f) {
      v16bf b0 = *(const v16bf*)&ks[(ln)      * KS + 32 * f + 16 * hi];
      v16bf b1 = *(const v16bf*)&ks[(16 + ln) * KS + 32 * f + 16 * hi];
      s0 = wmma_bf16(qf[f], b0, s0);
      s1 = wmma_bf16(qf[f], b1, s1);
    }

    // online softmax (row stats per lane; row = r + 8*hi, lanes 0..15 = cols)
    float corr[8];
#pragma unroll
    for (int r = 0; r < 8; ++r) {
      const int q = qbase + w * 16 + r + 8 * hi;
      float e0 = s0[r] * sc, e1 = s1[r] * sc;
      if (kb + ln      > q) e0 = -1e30f;
      if (kb + 16 + ln > q) e1 = -1e30f;
      float rm = fmaxf(e0, e1);
#pragma unroll
      for (int off = 1; off < 16; off <<= 1)
        rm = fmaxf(rm, __shfl_xor(rm, off, 32));
      const float mn = fmaxf(m[r], rm);
      const float p0 = __expf(e0 - mn), p1 = __expf(e1 - mn);
      float rs = p0 + p1;
#pragma unroll
      for (int off = 1; off < 16; off <<= 1)
        rs += __shfl_xor(rs, off, 32);
      corr[r] = __expf(m[r] - mn);
      ls[r] = ls[r] * corr[r] + rs;
      m[r] = mn;
      ps[(w * 16 + r + 8 * hi) * PS + ln]      = (__bf16)p0;
      ps[(w * 16 + r + 8 * hi) * PS + 16 + ln] = (__bf16)p1;
    }
#pragma unroll
    for (int nt = 0; nt < 12; ++nt)
#pragma unroll
      for (int r = 0; r < 8; ++r) o[nt][r] *= corr[r];

    // O += P . V   (P re-read as A-frag; same-wave DS ops are in order)
    v16bf pf = load_a_frag(&ps[(w * 16 + ln) * PS], hi);
#pragma unroll
    for (int nt = 0; nt < 12; ++nt) {
      v16bf bv = *(const v16bf*)&vt[(nt * 16 + ln) * VS + 16 * hi];
      o[nt] = wmma_bf16(pf, bv, o[nt]);
    }
    __syncthreads();
  }

  // normalize and store f32 output
  float rinv[8];
#pragma unroll
  for (int r = 0; r < 8; ++r) rinv[r] = 1.0f / ls[r];
#pragma unroll
  for (int nt = 0; nt < 12; ++nt)
#pragma unroll
    for (int r = 0; r < 8; ++r) {
      const int row = qr + r + 8 * hi;
      out[((size_t)b * TT + row) * HH + nt * 16 + ln] = o[nt][r] * rinv[r];
    }
}

// ---------------------------------------------------------------------------
extern "C" void kernel_launch(void* const* d_in, const int* in_sizes, int n_in,
                              void* d_out, int out_size, void* d_ws, size_t ws_size,
                              hipStream_t stream) {
  (void)in_sizes; (void)n_in; (void)out_size; (void)ws_size;
  const float* x  = (const float*)d_in[0];
  const float* Wq = (const float*)d_in[1];
  const float* Wk = (const float*)d_in[2];
  const float* Wv = (const float*)d_in[3];
  float* out = (float*)d_out;
  __bf16* ws = (__bf16*)d_ws;   // q | k | v, each NROW*HH bf16 (37.75 MB total)

  dim3 g1(NROW / 64, 3);
  qkv_rope_kernel<<<g1, 256, 0, stream>>>(x, Wq, Wk, Wv, ws);

  const __bf16* qws = ws;
  const __bf16* kws = ws + (size_t)NROW * HH;
  const __bf16* vws = ws + (size_t)2 * NROW * HH;
  dim3 g2(TT / 128, BB);
  flash_attn_kernel<<<g2, 256, 0, stream>>>(qws, kws, vws, out);
}
